// SelfAttention_6708738916864
// MI455X (gfx1250) — compile-verified, matrix operand order checked by reading, not run on hardware
//
#include <hip/hip_runtime.h>
#include <hip/hip_bf16.h>

#define B_ 4
#define S_ 2048
#define E_ 1024
#define H_ 16
#define D_ 64

typedef __attribute__((ext_vector_type(16))) __bf16 v16bf;
typedef __attribute__((ext_vector_type(8)))  float  v8f;

__device__ inline v8f wmma_bf16(v16bf a, v16bf b, v8f c) {
  // emits v_wmma_f32_16x16x32_bf16
  return __builtin_amdgcn_wmma_f32_16x16x32_bf16(false, a, false, b, (short)0, c,
                                                 false, false);
}

// A/B fragment gather from row-major storage where this lane owns one source
// row (A: matrix row m; B operand stored as W[n][k], lane owns row n).
// p points at (row_base + k0); kb = (lane>>4)*8 per ISA 16-bit A layout.
__device__ inline v16bf frag_row_bf16(const __bf16* p, int kb) {
  v16bf f;
#pragma unroll
  for (int j = 0; j < 8; ++j) {
    f[j]     = p[kb + j];
    f[j + 8] = p[kb + 16 + j];
  }
  return f;
}
__device__ inline v16bf frag_row_f32(const float* p, int kb) {
  v16bf f;
#pragma unroll
  for (int j = 0; j < 8; ++j) {
    f[j]     = (__bf16)p[kb + j];
    f[j + 8] = (__bf16)p[kb + 16 + j];
  }
  return f;
}
// B fragment from a row-major K x N matrix (lane owns column n), ld = N.
__device__ inline v16bf frag_col_bf16(const __bf16* p, int ld, int kb) {
  v16bf f;
#pragma unroll
  for (int j = 0; j < 8; ++j) {
    f[j]     = p[(kb + j) * ld];
    f[j + 8] = p[(kb + 16 + j) * ld];
  }
  return f;
}

// ---------------------------------------------------------------------------
// Kernel 1: per-head-dim QKV projection.  X viewed as [B*S*H, 64] row-major;
// one wave computes a 16x64 tile (= one token, all 16 heads).  Output stored
// bf16 as [B, H, S, D] for the attention kernel.
// ---------------------------------------------------------------------------
__global__ __launch_bounds__(256) void qkv_proj_kernel(
    const float* __restrict__ q, const float* __restrict__ k,
    const float* __restrict__ v, const float* __restrict__ Wq,
    const float* __restrict__ Wk, const float* __restrict__ Wv,
    __bf16* __restrict__ qp, __bf16* __restrict__ kp, __bf16* __restrict__ vp) {
  const float* X; const float* W; __bf16* O;
  if (blockIdx.y == 0)      { X = q; W = Wq; O = qp; }
  else if (blockIdx.y == 1) { X = k; W = Wk; O = kp; }
  else                      { X = v; W = Wv; O = vp; }

  const int wave = threadIdx.x >> 5;
  const int lane = threadIdx.x & 31;
  const int half = lane >> 4;
  const int ln   = lane & 15;
  const int kb   = half * 8;

  const int t = blockIdx.x * 8 + wave;  // token index in [0, B*S)
  const int b = t / S_;
  const int s = t - b * S_;

  v8f z = {};
  v8f acc[4];
#pragma unroll
  for (int nt = 0; nt < 4; ++nt) acc[nt] = z;

#pragma unroll
  for (int k0 = 0; k0 < 64; k0 += 32) {
    v16bf a = frag_row_f32(X + ((size_t)t * 16 + ln) * 64 + k0, kb);
#pragma unroll
    for (int nt = 0; nt < 4; ++nt) {
      // B[d][e] = W[e][d]; lane owns e = nt*16+ln, gathers along d.
      v16bf bf = frag_row_f32(W + (nt * 16 + ln) * 64 + k0, kb);
      acc[nt] = wmma_bf16(a, bf, acc[nt]);
    }
  }

#pragma unroll
  for (int nt = 0; nt < 4; ++nt)
#pragma unroll
    for (int r = 0; r < 8; ++r) {
      const int h = r + half * 8;  // C layout: row = vgpr + 8*half, col = ln
      O[(((size_t)b * 16 + h) * S_ + s) * 64 + nt * 16 + ln] =
          (__bf16)acc[nt][r];
    }
}

// ---------------------------------------------------------------------------
// Kernel 2: causal flash attention.  blockIdx.y = (b,h); 8 waves each own a
// 16-row query block; 32-key blocks staged into LDS with the CDNA5 async
// memory->LDS path (ASYNCcnt tracked); online softmax with f32 running
// stats; P re-laid out via LDS into an A fragment for P*V.
// ---------------------------------------------------------------------------
__global__ __launch_bounds__(256) void flash_attn_kernel(
    const __bf16* __restrict__ qp, const __bf16* __restrict__ kp,
    const __bf16* __restrict__ vp, __bf16* __restrict__ ao) {
  __shared__ __bf16 Kt[32 * 64];
  __shared__ __bf16 Vt[32 * 64];
  __shared__ __bf16 Pt[8][16 * 32];

  const int tid  = threadIdx.x;
  const int wave = tid >> 5;
  const int lane = tid & 31;
  const int half = lane >> 4;
  const int ln   = lane & 15;
  const int kb   = half * 8;

  const int bh = blockIdx.y;
  const int b  = bh >> 4;
  const int h  = bh & 15;
  const int q0 = (blockIdx.x * 8 + wave) * 16;

  const __bf16* Q = qp + (size_t)bh * S_ * 64;
  const __bf16* K = kp + (size_t)bh * S_ * 64;
  const __bf16* V = vp + (size_t)bh * S_ * 64;

  // LDS byte offsets for the async copies (low 32 bits of the generic
  // pointer to a __shared__ object == wave-relative LDS address).
  const unsigned ldsK = (unsigned)(uintptr_t)(&Kt[0]) + (unsigned)tid * 16u;
  const unsigned ldsV = (unsigned)(uintptr_t)(&Vt[0]) + (unsigned)tid * 16u;

  // Q fragments (16x64 = two 16x32 A fragments), kept resident.
  v16bf qa0 = frag_row_bf16(Q + ((size_t)q0 + ln) * 64 + 0,  kb);
  v16bf qa1 = frag_row_bf16(Q + ((size_t)q0 + ln) * 64 + 32, kb);

  float m_i[8], l_i[8];
  v8f z = {};
  v8f o[4];
#pragma unroll
  for (int r = 0; r < 8; ++r) { m_i[r] = -3.0e38f; l_i[r] = 0.0f; }
#pragma unroll
  for (int nt = 0; nt < 4; ++nt) o[nt] = z;

  const int jend = (blockIdx.x + 1) * 128;  // max keys needed by this block
  for (int j0 = 0; j0 < jend; j0 += 32) {
    __syncthreads();
    // Async stage: 32 keys x 64 dims bf16 = 4 KB contiguous per tile,
    // 256 threads x 16 B each, memory -> LDS without a VGPR round trip.
    {
      const __bf16* gk = K + (size_t)j0 * 64 + tid * 8;
      const __bf16* gv = V + (size_t)j0 * 64 + tid * 8;
      asm volatile(
          "global_load_async_to_lds_b128 %0, %1, off\n\t"
          "global_load_async_to_lds_b128 %2, %3, off"
          :
          : "v"(ldsK), "v"(gk), "v"(ldsV), "v"(gv)
          : "memory");
    }
    if (j0 + 32 < jend) {  // global_prefetch_b8 of the next K/V tiles
      __builtin_prefetch((const void*)(K + (size_t)(j0 + 32) * 64 + tid * 8), 0, 0);
      __builtin_prefetch((const void*)(V + (size_t)(j0 + 32) * 64 + tid * 8), 0, 0);
    }
    asm volatile("s_wait_asynccnt 0" ::: "memory");
    __syncthreads();

    if (j0 <= q0 + 15) {  // wave-uniform: EXEC stays all-ones for WMMA
      v8f sa = z, sb = z;  // S tile cols [0,16) and [16,32)
#pragma unroll
      for (int k0 = 0; k0 < 64; k0 += 32) {
        v16bf a   = (k0 == 0) ? qa0 : qa1;
        v16bf bk0 = frag_row_bf16(&Kt[ln * 64 + k0], kb);         // key j0+ln
        v16bf bk1 = frag_row_bf16(&Kt[(16 + ln) * 64 + k0], kb);  // key j0+16+ln
        sa = wmma_bf16(a, bk0, sa);
        sb = wmma_bf16(a, bk1, sb);
      }
      float corr[8];
#pragma unroll
      for (int r = 0; r < 8; ++r) {
        const int qrow = q0 + r + half * 8;
        float xa = sa[r] * 0.125f;  // 1/sqrt(64)
        float xb = sb[r] * 0.125f;
        if (j0 + ln      > qrow) xa = -1.0e30f;  // causal mask
        if (j0 + 16 + ln > qrow) xb = -1.0e30f;
        float rmax = fmaxf(xa, xb);
#pragma unroll
        for (int m = 1; m < 16; m <<= 1)
          rmax = fmaxf(rmax, __shfl_xor(rmax, m, 32));
        const float newm = fmaxf(m_i[r], rmax);
        const float pa = __expf(xa - newm);
        const float pb = __expf(xb - newm);
        float rsum = pa + pb;
#pragma unroll
        for (int m = 1; m < 16; m <<= 1) rsum += __shfl_xor(rsum, m, 32);
        const float c = __expf(m_i[r] - newm);
        l_i[r]  = l_i[r] * c + rsum;
        m_i[r]  = newm;
        corr[r] = c;
        // C layout -> row-major P in LDS (per-wave private region)
        Pt[wave][(r + half * 8) * 32 + ln]      = (__bf16)pa;
        Pt[wave][(r + half * 8) * 32 + 16 + ln] = (__bf16)pb;
      }
      // P as a 16x32 A fragment; V tile as 32x16 B fragments.
      v16bf pfrag = frag_row_bf16(&Pt[wave][ln * 32], kb);
#pragma unroll
      for (int nt = 0; nt < 4; ++nt) {
#pragma unroll
        for (int r = 0; r < 8; ++r) o[nt][r] = o[nt][r] * corr[r];
        v16bf vb = frag_col_bf16(&Vt[nt * 16 + ln], 64, kb);
        o[nt] = wmma_bf16(pfrag, vb, o[nt]);
      }
    }
  }

  // Normalize and store attention output bf16 as [B, S, E] (e = h*64 + d).
#pragma unroll
  for (int r = 0; r < 8; ++r) {
    const float inv  = 1.0f / l_i[r];
    const int   qrow = q0 + r + half * 8;
#pragma unroll
    for (int nt = 0; nt < 4; ++nt)
      ao[((size_t)b * S_ + qrow) * E_ + h * 64 + nt * 16 + ln] =
          (__bf16)(o[nt][r] * inv);
  }
}

// ---------------------------------------------------------------------------
// Kernel 3: output projection  Y = AO @ Wo^T + bo,  [8192 x 1024]x[1024 x 1024].
// One wave computes a 16x64 tile; K loop of 32 (32 WMMAs per 16x16 tile).
// ---------------------------------------------------------------------------
__global__ __launch_bounds__(256) void out_proj_kernel(
    const __bf16* __restrict__ ao, const float* __restrict__ Wo,
    const float* __restrict__ bo, float* __restrict__ y) {
  const int wave = threadIdx.x >> 5;
  const int lane = threadIdx.x & 31;
  const int half = lane >> 4;
  const int ln   = lane & 15;
  const int kb   = half * 8;

  const int w  = blockIdx.x * 8 + wave;  // [0, 8192)
  const int mt = w >> 4;                 // 16-row (token) tile
  const int ng = w & 15;                 // 64-column group

  v8f z = {};
  v8f acc[4];
#pragma unroll
  for (int nt = 0; nt < 4; ++nt) acc[nt] = z;

  for (int k0 = 0; k0 < E_; k0 += 32) {
    v16bf a = frag_row_bf16(ao + ((size_t)mt * 16 + ln) * E_ + k0, kb);
#pragma unroll
    for (int nt = 0; nt < 4; ++nt) {
      // B[c][e] = Wo[e][c]; lane owns e, gathers row e of Wo along c.
      v16bf bf =
          frag_row_f32(Wo + (size_t)(ng * 64 + nt * 16 + ln) * E_ + k0, kb);
      acc[nt] = wmma_bf16(a, bf, acc[nt]);
    }
  }

#pragma unroll
  for (int nt = 0; nt < 4; ++nt) {
    const int   e    = ng * 64 + nt * 16 + ln;
    const float bias = bo[e];
#pragma unroll
    for (int r = 0; r < 8; ++r) {
      const int t = mt * 16 + r + half * 8;
      y[(size_t)t * E_ + e] = acc[nt][r] + bias;
    }
  }
}

// ---------------------------------------------------------------------------
extern "C" void kernel_launch(void* const* d_in, const int* in_sizes, int n_in,
                              void* d_out, int out_size, void* d_ws,
                              size_t ws_size, hipStream_t stream) {
  (void)in_sizes; (void)n_in; (void)out_size; (void)ws_size;
  const float* q  = (const float*)d_in[0];
  const float* k  = (const float*)d_in[1];
  const float* v  = (const float*)d_in[2];
  // d_in[3] is the causal mask; causality is applied analytically.
  const float* Wq = (const float*)d_in[4];
  const float* Wk = (const float*)d_in[5];
  const float* Wv = (const float*)d_in[6];
  const float* Wo = (const float*)d_in[7];
  const float* bo = (const float*)d_in[8];

  const size_t nPerTensor = (size_t)B_ * H_ * S_ * D_;  // 8,388,608 elems
  __bf16* qp = (__bf16*)d_ws;
  __bf16* kp = qp + nPerTensor;
  __bf16* vp = kp + nPerTensor;
  __bf16* ao = vp + nPerTensor;  // total 64 MB bf16 workspace

  qkv_proj_kernel<<<dim3(1024, 3), 256, 0, stream>>>(q, k, v, Wq, Wk, Wv,
                                                     qp, kp, vp);
  flash_attn_kernel<<<dim3(16, 64), 256, 0, stream>>>(qp, kp, vp, ao);
  out_proj_kernel<<<dim3(1024), 256, 0, stream>>>(ao, Wo, bo, (float*)d_out);
}
